// InteractionPPBlock_47115791237975
// MI455X (gfx1250) — compile-verified, matrix operand order checked by reading, not run on hardware
//
#include <hip/hip_runtime.h>
#include <hip/hip_bf16.h>

// ---------------------------------------------------------------------------
// DimeNet++ InteractionPPBlock for MI455X (gfx1250, wave32, WMMA).
// All GEMMs via v_wmma_f32_16x16x32_f16 (f16 inputs, f32 accumulate).
// Memory-bound on the triplet path -> fuse everything T-sized; keep h-tile
// resident in LDS through the whole residual chain in the fuse kernel.
// ---------------------------------------------------------------------------

typedef __attribute__((ext_vector_type(16))) _Float16 v16h;
typedef __attribute__((ext_vector_type(8)))  float    v8f;

// Fast silu: x * rcp(1+exp(-x)).  v_exp_f32 + v_add + v_rcp_f32 + v_mul.
__device__ __forceinline__ float silu_f(float v) {
    return v * __builtin_amdgcn_rcpf(1.0f + __expf(-v));
}

// Build a 16x32 f16 fragment (A or B) from an LDS tile stored row-major over K
// ([row][k], stride ld).  Per ISA 7.12.2: lane l holds row (l&15); K pattern:
//   VGPR v (pair) -> K = 2*(v&3) + 16*(v>>2) + 8*(l>=16)  (+0/+1 within pair)
__device__ __forceinline__ v16h frag_ld(const _Float16* base, int ld) {
    const int lane = threadIdx.x & 31;
    const _Float16* row = base + (lane & 15) * ld + ((lane >> 4) << 3);
    v16h f;
#pragma unroll
    for (int v = 0; v < 8; ++v) {
        const int k = ((v & 3) << 1) + ((v >> 2) << 4);
        f[2 * v]     = row[k];
        f[2 * v + 1] = row[k + 1];
    }
    return f;
}

// One 16x16 output tile: A is [16 rows][K] at lda, B (weightsT) is [16 cols][K] at ldb.
__device__ __forceinline__ v8f wmma_tile(const _Float16* A, int lda,
                                         const _Float16* B, int ldb, int K) {
    v8f acc = {};
#pragma unroll
    for (int k = 0; k < 128; k += 32) {
        if (k >= K) break;
        v16h a = frag_ld(A + k, lda);
        v16h b = frag_ld(B + k, ldb);
        acc = __builtin_amdgcn_wmma_f32_16x16x32_f16(
            false, a, false, b, (short)0, acc, false, false);
    }
    return acc;
}

// D tile element (r in 0..7): row = r + 8*(lane>=16), col = lane&15
__device__ __forceinline__ int out_row(int r) {
    return r + (((threadIdx.x & 31) >> 4) << 3);
}
__device__ __forceinline__ int out_col() { return threadIdx.x & 15; }

// Load 128 rows x KSRC f32 from global into f16 LDS [m][k] (stride ld),
// zero-padding rows >= nrows and cols >= KSRC (up to KPAD).
template <int KSRC, int KPAD>
__device__ __forceinline__ void load_rows_f16(const float* __restrict__ src,
                                              long row0, long nrows,
                                              _Float16* dst, int ld) {
    for (int i = threadIdx.x; i < 128 * KPAD; i += (int)blockDim.x) {
        const int m = i / KPAD, k = i % KPAD;
        const long r = row0 + m;
        float v = 0.0f;
        if (r < nrows && k < KSRC) v = src[r * (long)KSRC + k];
        dst[m * ld + k] = (_Float16)v;
    }
}

// Load a KxN f32 weight from global, transposed into f16 LDS [n][k] (stride ld).
template <int K, int N>
__device__ __forceinline__ void load_weight_T(const float* __restrict__ W,
                                              _Float16* dst, int ld) {
    for (int i = threadIdx.x; i < K * N; i += (int)blockDim.x) {
        const int k = i / N, n = i % N;
        dst[n * ld + k] = (_Float16)W[i];
    }
}

// ---------------------------------------------------------------------------
__global__ void ipp_zero_kernel(float* __restrict__ p, long n) {
    long i = (long)blockIdx.x * blockDim.x + threadIdx.x;
    if (i < n) p[i] = 0.0f;
}

// Wc[42][64] = W_sbf1[42][8] @ W_sbf2[8][64]
__global__ void ipp_wc_kernel(const float* __restrict__ W_sbf1,
                              const float* __restrict__ W_sbf2,
                              float* __restrict__ Wc) {
    int i = blockIdx.x * blockDim.x + threadIdx.x;
    if (i < 42 * 64) {
        const int r = i / 64, c = i % 64;
        float s = 0.0f;
#pragma unroll
        for (int b = 0; b < 8; ++b) s += W_sbf1[r * 8 + b] * W_sbf2[b * 64 + c];
        Wc[i] = s;
    }
}

// ---------------------------------------------------------------------------
// Edge preprocessing: x_ji = silu(x@W_ji+b); x_kj = silu(x@W_kj+b)*rbfmod;
// xkjd = silu(x_kj @ W_down).  128 edges per block, 8 waves.
// ---------------------------------------------------------------------------
__global__ __launch_bounds__(256) void ipp_edge_pre_kernel(
    const float* __restrict__ x, const float* __restrict__ rbf,
    const float* __restrict__ W_ji, const float* __restrict__ b_ji,
    const float* __restrict__ W_kj, const float* __restrict__ b_kj,
    const float* __restrict__ W_rbf1, const float* __restrict__ W_rbf2,
    const float* __restrict__ W_down,
    float* __restrict__ x_ji_out, float* __restrict__ xkjd_out, int E) {
    __shared__ _Float16 xH[128 * 136];   // x tile (f16) [m][k]
    __shared__ _Float16 wgt[128 * 136];  // current weight, transposed [n][k]
    __shared__ _Float16 tH[128 * 136];   // x_kj (f16) [m][k]
    __shared__ float rbfT[128 * 8];      // per-row rbf@W_rbf1
    __shared__ float Wr2[8 * 128];       // W_rbf2
    __shared__ float biasS[128];

    const int tid = threadIdx.x;
    const int wave = tid >> 5;
    const long e0 = (long)blockIdx.x * 128;
    const int rows = (E - (int)e0) < 128 ? (E - (int)e0) : 128;

    load_rows_f16<128, 128>(x, e0, E, xH, 136);
    for (int i = tid; i < 8 * 128; i += 256) Wr2[i] = W_rbf2[i];
    if (tid < 128) {
        float t[8] = {0, 0, 0, 0, 0, 0, 0, 0};
        if (tid < rows) {
            const long e = e0 + tid;
            float rv[6];
#pragma unroll
            for (int j = 0; j < 6; ++j) rv[j] = rbf[e * 6 + j];
#pragma unroll
            for (int b = 0; b < 8; ++b) {
                float s = 0.0f;
#pragma unroll
                for (int j = 0; j < 6; ++j) s += rv[j] * W_rbf1[j * 8 + b];
                t[b] = s;
            }
        }
#pragma unroll
        for (int b = 0; b < 8; ++b) rbfT[tid * 8 + b] = t[b];
    }
    load_weight_T<128, 128>(W_ji, wgt, 136);
    if (tid < 128) biasS[tid] = b_ji[tid];
    __syncthreads();

    // GEMM1: x_ji = silu(x@W_ji + b_ji) -> global
    for (int tt = wave; tt < 64; tt += 8) {
        const int mt = tt >> 3, nt = tt & 7;
        v8f acc = wmma_tile(&xH[mt * 16 * 136], 136, &wgt[nt * 16 * 136], 136, 128);
        const int n = nt * 16 + out_col();
        const float bn = biasS[n];
#pragma unroll
        for (int r = 0; r < 8; ++r) {
            const int m = mt * 16 + out_row(r);
            const float vres = silu_f(acc[r] + bn);
            if (m < rows) x_ji_out[(e0 + m) * 128 + n] = vres;
        }
    }
    __syncthreads();
    load_weight_T<128, 128>(W_kj, wgt, 136);
    if (tid < 128) biasS[tid] = b_kj[tid];
    __syncthreads();

    // GEMM2: x_kj = silu(x@W_kj + b_kj) * (rbf@W_rbf1@W_rbf2) -> tH (f16)
    for (int tt = wave; tt < 64; tt += 8) {
        const int mt = tt >> 3, nt = tt & 7;
        v8f acc = wmma_tile(&xH[mt * 16 * 136], 136, &wgt[nt * 16 * 136], 136, 128);
        const int n = nt * 16 + out_col();
        const float bn = biasS[n];
        float w2[8];
#pragma unroll
        for (int b = 0; b < 8; ++b) w2[b] = Wr2[b * 128 + n];
#pragma unroll
        for (int r = 0; r < 8; ++r) {
            const int m = mt * 16 + out_row(r);
            const float v = silu_f(acc[r] + bn);
            float mod = 0.0f;
#pragma unroll
            for (int b = 0; b < 8; ++b) mod += rbfT[m * 8 + b] * w2[b];
            tH[m * 136 + n] = (_Float16)(v * mod);  // pad rows: mod==0 -> 0
        }
    }
    __syncthreads();
    load_weight_T<128, 64>(W_down, wgt, 136);
    __syncthreads();

    // GEMM3: xkjd = silu(x_kj @ W_down)  [128 x 64]
    for (int tt = wave; tt < 32; tt += 8) {
        const int mt = tt >> 2, nt = tt & 3;
        v8f acc = wmma_tile(&tH[mt * 16 * 136], 136, &wgt[nt * 16 * 136], 136, 128);
        const int n = nt * 16 + out_col();
#pragma unroll
        for (int r = 0; r < 8; ++r) {
            const int m = mt * 16 + out_row(r);
            const float vres = silu_f(acc[r]);
            if (m < rows) xkjd_out[(e0 + m) * 64 + n] = vres;
        }
    }
}

// ---------------------------------------------------------------------------
// Triplet path: s = sbf@Wc; xk = xkjd[idx_kj]; y = xk@sel_W[sel(angle)];
// atomically accumulate (y*s) into agg[idx_ji].  128 triplets per block.
// MoE routing: compute all 4 branch WMMAs, select per-lane (v_cndmask).
// ---------------------------------------------------------------------------
__global__ __launch_bounds__(256) void ipp_triplet_kernel(
    const float* __restrict__ sbf, const float* __restrict__ angle,
    const int* __restrict__ idx_kj, const int* __restrict__ idx_ji,
    const float* __restrict__ Wc, const float* __restrict__ selW,
    const float* __restrict__ xkjd, float* __restrict__ agg, long T) {
    __shared__ _Float16 sbH[128 * 72];      // sbf tile, K padded 42->64
    __shared__ _Float16 wcT[64 * 72];       // Wc transposed [n][k], zero pad
    __shared__ _Float16 xkH[128 * 72];      // gathered xk rows (f16)
    __shared__ _Float16 swT[4 * 64 * 72];   // sel_W[c] transposed [n][k]
    __shared__ float    sF[128 * 68];       // s (f32)
    __shared__ int selv[128];
    __shared__ int jiArr[128];

    const int tid = threadIdx.x;
    const int wave = tid >> 5;
    const long t0 = (long)blockIdx.x * 128;
    const long rem = T - t0;
    const int rowsT = rem < 128 ? (int)rem : 128;

    for (int i = tid; i < 128 * 64; i += 256) {
        const int m = i >> 6, k = i & 63;
        float v = (m < rowsT && k < 42) ? sbf[(t0 + m) * 42 + k] : 0.0f;
        sbH[m * 72 + k] = (_Float16)v;
    }
    for (int i = tid; i < 64 * 64; i += 256) {
        const int k = i >> 6, n = i & 63;
        wcT[n * 72 + k] = (_Float16)((k < 42) ? Wc[k * 64 + n] : 0.0f);
    }
    for (int i = tid; i < 4 * 64 * 64; i += 256) {
        const int c = i >> 12, k = (i >> 6) & 63, n = i & 63;
        swT[c * 64 * 72 + n * 72 + k] = (_Float16)selW[i];
    }
    if (tid < 128) {
        int c = 0, e = 0;
        if (tid < rowsT) {
            const float a = angle[t0 + tid];
            c = (int)floorf(a * (4.0f / 3.141593f));
            c = c < 0 ? 0 : (c > 3 ? 3 : c);
            e = idx_ji[t0 + tid];
        }
        selv[tid] = c;
        jiArr[tid] = e;
    }
    // gather xk rows: 2 threads per row, 8 float4 (global_load_b128) each
    {
        const int m = tid >> 1;
        const int half = tid & 1;
        const bool inb = m < rowsT;
        const int ek = inb ? idx_kj[t0 + m] : 0;
        const float4* src =
            (const float4*)(xkjd + (long)ek * 64 + half * 32);
        _Float16* dst = &xkH[m * 72 + half * 32];
#pragma unroll
        for (int q = 0; q < 8; ++q) {
            float4 v = make_float4(0.f, 0.f, 0.f, 0.f);
            if (inb) v = src[q];
            dst[4 * q + 0] = (_Float16)v.x;
            dst[4 * q + 1] = (_Float16)v.y;
            dst[4 * q + 2] = (_Float16)v.z;
            dst[4 * q + 3] = (_Float16)v.w;
        }
    }
    __syncthreads();

    // s = sbf @ Wc (K=64, upper 22 K zero-padded)
    for (int tt = wave; tt < 32; tt += 8) {
        const int mt = tt >> 2, nt = tt & 3;
        v8f acc = wmma_tile(&sbH[mt * 16 * 72], 72, &wcT[nt * 16 * 72], 72, 64);
#pragma unroll
        for (int r = 0; r < 8; ++r)
            sF[(mt * 16 + out_row(r)) * 68 + nt * 16 + out_col()] = acc[r];
    }
    __syncthreads();

    // routed matmul + scale + segment-sum scatter
    for (int tt = wave; tt < 32; tt += 8) {
        const int mt = tt >> 2, nt = tt & 3;
        const _Float16* A = &xkH[mt * 16 * 72];
        v8f a0 = wmma_tile(A, 72, &swT[0 * 64 * 72 + nt * 16 * 72], 72, 64);
        v8f a1 = wmma_tile(A, 72, &swT[1 * 64 * 72 + nt * 16 * 72], 72, 64);
        v8f a2 = wmma_tile(A, 72, &swT[2 * 64 * 72 + nt * 16 * 72], 72, 64);
        v8f a3 = wmma_tile(A, 72, &swT[3 * 64 * 72 + nt * 16 * 72], 72, 64);
        const int n = nt * 16 + out_col();
#pragma unroll
        for (int r = 0; r < 8; ++r) {
            const int m = mt * 16 + out_row(r);
            const int c = selv[m];
            const float y =
                (c == 0) ? a0[r] : (c == 1) ? a1[r] : (c == 2) ? a2[r] : a3[r];
            const float val = y * sF[m * 68 + n];
            if (m < rowsT) {
                __hip_atomic_fetch_add(&agg[(long)jiArr[m] * 64 + n], val,
                                       __ATOMIC_RELAXED, __HIP_MEMORY_SCOPE_AGENT);
            }
        }
    }
}

// ---------------------------------------------------------------------------
// Fused output chain: h = x_ji + silu(agg@W_up); 1x before-skip residual MLP;
// h = silu(h@W_lin+b)+x; 2x after-skip residual MLP; write h.
// h kept as f32 master tile in LDS with an f16 shadow for WMMA A-fragments.
// ---------------------------------------------------------------------------
__device__ __forceinline__ void gemm_silu_to(const _Float16* Ash, const _Float16* Wsh,
                                             const float* biasSh, _Float16* dstH,
                                             int wave) {
    for (int tt = wave; tt < 64; tt += 8) {
        const int mt = tt >> 3, nt = tt & 7;
        v8f acc = wmma_tile(Ash + mt * 16 * 136, 136, Wsh + nt * 16 * 136, 136, 128);
        const int n = nt * 16 + out_col();
        const float bn = biasSh[n];
#pragma unroll
        for (int r = 0; r < 8; ++r) {
            const int m = mt * 16 + out_row(r);
            dstH[m * 136 + n] = (_Float16)silu_f(acc[r] + bn);
        }
    }
}

__device__ __forceinline__ void gemm_silu_addto(const _Float16* Ash, const _Float16* Wsh,
                                                const float* biasSh, float* hFsh,
                                                int wave) {
    for (int tt = wave; tt < 64; tt += 8) {
        const int mt = tt >> 3, nt = tt & 7;
        v8f acc = wmma_tile(Ash + mt * 16 * 136, 136, Wsh + nt * 16 * 136, 136, 128);
        const int n = nt * 16 + out_col();
        const float bn = biasSh[n];
#pragma unroll
        for (int r = 0; r < 8; ++r) {
            const int m = mt * 16 + out_row(r);
            hFsh[m * 132 + n] += silu_f(acc[r] + bn);
        }
    }
}

__global__ __launch_bounds__(256) void ipp_fuse_kernel(
    const float* __restrict__ agg, const float* __restrict__ x_ji,
    const float* __restrict__ x, const float* __restrict__ W_up,
    const float* __restrict__ Wb1, const float* __restrict__ bb1,
    const float* __restrict__ Wb2, const float* __restrict__ bb2,
    const float* __restrict__ W_lin, const float* __restrict__ b_lin,
    const float* __restrict__ Wa1, const float* __restrict__ ba1,
    const float* __restrict__ Wa2, const float* __restrict__ ba2,
    float* __restrict__ out, int E) {
    __shared__ float    hF[128 * 132];   // f32 master h tile
    __shared__ _Float16 hH[128 * 136];   // f16 shadow of h (WMMA A)
    __shared__ _Float16 tH[128 * 136];   // inner-MLP activation
    __shared__ _Float16 wgt[128 * 136];  // current weight transposed
    __shared__ float biasS[128];

    const int tid = threadIdx.x;
    const int wave = tid >> 5;
    const long e0 = (long)blockIdx.x * 128;
    const int rows = (E - (int)e0) < 128 ? (E - (int)e0) : 128;

    // P0: h = x_ji + silu(agg @ W_up)
    for (int i = tid; i < 128 * 64; i += 256) {
        const int m = i >> 6, k = i & 63;
        tH[m * 136 + k] = (_Float16)((m < rows) ? agg[(e0 + m) * 64 + k] : 0.0f);
    }
    load_weight_T<64, 128>(W_up, wgt, 136);
    __syncthreads();
    for (int tt = wave; tt < 64; tt += 8) {
        const int mt = tt >> 3, nt = tt & 7;
        v8f acc = wmma_tile(&tH[mt * 16 * 136], 136, &wgt[nt * 16 * 136], 136, 64);
        const int n = nt * 16 + out_col();
#pragma unroll
        for (int r = 0; r < 8; ++r) {
            const int m = mt * 16 + out_row(r);
            const float u = silu_f(acc[r]);
            float h = 0.0f;
            if (m < rows) h = x_ji[(e0 + m) * 128 + n] + u;
            hF[m * 132 + n] = h;
            hH[m * 136 + n] = (_Float16)h;
        }
    }
    __syncthreads();

    // before-skip residual MLP (NB=1)
    load_weight_T<128, 128>(Wb1, wgt, 136);
    if (tid < 128) biasS[tid] = bb1[tid];
    __syncthreads();
    gemm_silu_to(hH, wgt, biasS, tH, wave);
    __syncthreads();
    load_weight_T<128, 128>(Wb2, wgt, 136);
    if (tid < 128) biasS[tid] = bb2[tid];
    __syncthreads();
    gemm_silu_addto(tH, wgt, biasS, hF, wave);
    __syncthreads();

    // h = silu(h @ W_lin + b_lin) + x
    for (int i = tid; i < 128 * 128; i += 256) {
        const int m = i >> 7, n = i & 127;
        hH[m * 136 + n] = (_Float16)hF[m * 132 + n];
    }
    load_weight_T<128, 128>(W_lin, wgt, 136);
    if (tid < 128) biasS[tid] = b_lin[tid];
    __syncthreads();
    for (int tt = wave; tt < 64; tt += 8) {
        const int mt = tt >> 3, nt = tt & 7;
        v8f acc = wmma_tile(&hH[mt * 16 * 136], 136, &wgt[nt * 16 * 136], 136, 128);
        const int n = nt * 16 + out_col();
        const float bn = biasS[n];
#pragma unroll
        for (int r = 0; r < 8; ++r) {
            const int m = mt * 16 + out_row(r);
            float h = silu_f(acc[r] + bn);
            if (m < rows) h += x[(e0 + m) * 128 + n];
            hF[m * 132 + n] = h;
        }
    }
    __syncthreads();

    // after-skip residual MLPs (NA=2)
    for (int i2 = 0; i2 < 2; ++i2) {
        for (int i = tid; i < 128 * 128; i += 256) {
            const int m = i >> 7, n = i & 127;
            hH[m * 136 + n] = (_Float16)hF[m * 132 + n];
        }
        load_weight_T<128, 128>(Wa1 + i2 * 128 * 128, wgt, 136);
        if (tid < 128) biasS[tid] = ba1[i2 * 128 + tid];
        __syncthreads();
        gemm_silu_to(hH, wgt, biasS, tH, wave);
        __syncthreads();
        load_weight_T<128, 128>(Wa2 + i2 * 128 * 128, wgt, 136);
        if (tid < 128) biasS[tid] = ba2[i2 * 128 + tid];
        __syncthreads();
        gemm_silu_addto(tH, wgt, biasS, hF, wave);
        __syncthreads();
    }

    for (int i = tid; i < 128 * 128; i += 256) {
        const int m = i >> 7, n = i & 127;
        if (m < rows) out[(e0 + m) * 128 + n] = hF[m * 132 + n];
    }
}

// ---------------------------------------------------------------------------
extern "C" void kernel_launch(void* const* d_in, const int* in_sizes, int n_in,
                              void* d_out, int out_size, void* d_ws, size_t ws_size,
                              hipStream_t stream) {
    const float* x      = (const float*)d_in[0];
    const float* rbf    = (const float*)d_in[1];
    const float* sbf    = (const float*)d_in[2];
    const float* angle  = (const float*)d_in[3];
    const float* W_ji   = (const float*)d_in[4];
    const float* b_ji   = (const float*)d_in[5];
    const float* W_kj   = (const float*)d_in[6];
    const float* b_kj   = (const float*)d_in[7];
    const float* W_rbf1 = (const float*)d_in[8];
    const float* W_rbf2 = (const float*)d_in[9];
    const float* W_sbf1 = (const float*)d_in[10];
    const float* W_sbf2 = (const float*)d_in[11];
    const float* W_down = (const float*)d_in[12];
    const float* W_up   = (const float*)d_in[13];
    const float* selW   = (const float*)d_in[14];
    const float* Wb1    = (const float*)d_in[15];
    const float* bb1    = (const float*)d_in[16];
    const float* Wb2    = (const float*)d_in[17];
    const float* bb2    = (const float*)d_in[18];
    const float* W_lin  = (const float*)d_in[19];
    const float* b_lin  = (const float*)d_in[20];
    const float* Wa1    = (const float*)d_in[21];
    const float* ba1    = (const float*)d_in[22];
    const float* Wa2    = (const float*)d_in[23];
    const float* ba2    = (const float*)d_in[24];
    const int* idx_kj   = (const int*)d_in[25];
    const int* idx_ji   = (const int*)d_in[26];

    const int  E = in_sizes[0] / 128;
    const long T = (long)in_sizes[3];

    float* ws    = (float*)d_ws;
    float* x_ji  = ws;                           // E*128
    float* xkjd  = x_ji + (size_t)E * 128;       // E*64
    float* agg   = xkjd + (size_t)E * 64;        // E*64
    float* Wc    = agg + (size_t)E * 64;         // 42*64

    const long aggN = (long)E * 64;
    ipp_zero_kernel<<<(int)((aggN + 1023) / 1024), 1024, 0, stream>>>(agg, aggN);
    ipp_wc_kernel<<<(42 * 64 + 255) / 256, 256, 0, stream>>>(W_sbf1, W_sbf2, Wc);

    const int eb = (E + 127) / 128;
    ipp_edge_pre_kernel<<<eb, 256, 0, stream>>>(x, rbf, W_ji, b_ji, W_kj, b_kj,
                                                W_rbf1, W_rbf2, W_down,
                                                x_ji, xkjd, E);
    const int tb = (int)((T + 127) / 128);
    ipp_triplet_kernel<<<tb, 256, 0, stream>>>(sbf, angle, idx_kj, idx_ji,
                                               Wc, selW, xkjd, agg, T);
    ipp_fuse_kernel<<<eb, 256, 0, stream>>>(agg, x_ji, x, W_up, Wb1, bb1, Wb2, bb2,
                                            W_lin, b_lin, Wa1, ba1, Wa2, ba2,
                                            (float*)d_out, E);
}
